// GATModel_34780645163110
// MI455X (gfx1250) — matrix operand order repaired and checked by reading, not compiled, not run on hardware
//
#include <hip/hip_runtime.h>
#include <hip/hip_bf16.h>
#include <math.h>

typedef __bf16 v16bf __attribute__((ext_vector_type(16)));
typedef __bf16 v8bf  __attribute__((ext_vector_type(8)));
typedef float  v8f   __attribute__((ext_vector_type(8)));

#define NEG_SLOPE 0.2f

static constexpr int NN  = 8192;     // nodes
static constexpr int NE  = 32768;    // edges (without self loops)
static constexpr int ET  = NN + NE;  // edges + self loops = 40960
static constexpr int DD  = 768;
static constexpr int H1  = 8;
static constexpr int HC1 = H1 * DD;  // 6144
static constexpr unsigned ORD_NEG_INF = 0x007FFFFFu; // f2ord(-inf)

// ---------------- order-preserving float <-> uint (for atomic max) ---------
__device__ __forceinline__ unsigned f2ord(float f) {
  unsigned u = __float_as_uint(f);
  return (u & 0x80000000u) ? ~u : (u | 0x80000000u);
}
__device__ __forceinline__ float ord2f(unsigned u) {
  unsigned b = (u & 0x80000000u) ? (u ^ 0x80000000u) : ~u;
  return __uint_as_float(b);
}

// ---------------- WMMA bf16 GEMM: C[M,N] = A[M,K] @ B[K,N] (f32 in memory) -
// 128x128 block tile, 8 waves (4x2), each wave 32x64 (2x4 WMMA 16x16 tiles).
// LDS layouts: lA[row][k] and lBt[n][k] (B transposed) so every WMMA fragment
// is two contiguous 16B runs -> ds_load_b128. 40-element row pitch keeps 16B
// alignment and spreads 16-lane fragment reads across banks (stride 20 banks).
#define TM 128
#define TN 128
#define TK 32
#define LPAD 40

__global__ void __launch_bounds__(256)
gemm_bf16_wmma(const float* __restrict__ A, const float* __restrict__ B,
               float* __restrict__ C, int M, int N, int K) {
  __shared__ __bf16 lA [TM * LPAD];   // [row][k]
  __shared__ __bf16 lBt[TN * LPAD];   // [n][k]  (transposed B tile)

  const int tid  = threadIdx.x;
  const int lane = tid & 31;
  const int wid  = tid >> 5;       // 0..7
  const int wm   = wid >> 1;       // 0..3 -> +32 rows each
  const int wn   = wid & 1;        // 0..1 -> +64 cols each
  const int half = lane >> 4;      // 0/1
  const int l16  = lane & 15;
  const int rowBase = blockIdx.y * TM;
  const int colBase = blockIdx.x * TN;

  // fixed per-thread loader assignments (2 chunks of 8 elements each)
  int ar[2], acg[2], bn[2], bkg[2];
  const float* pA[2];
  const float* pB[2];
#pragma unroll
  for (int i = 0; i < 2; ++i) {
    int lin = tid + i * 256;
    ar[i]  = lin >> 2;           // A row 0..127
    acg[i] = (lin & 3) * 8;      // A k-group {0,8,16,24}
    bn[i]  = lin & 127;          // B col 0..127 (coalesced across lanes)
    bkg[i] = (lin >> 7) * 8;     // B k-group {0,8,16,24}
    pA[i] = A + (size_t)(rowBase + ar[i]) * K + acg[i];
    pB[i] = B + (size_t)bkg[i] * N + colBase + bn[i];
  }

  float aReg[2][8], bReg[2][8];
  // prologue: load first K tile into registers
#pragma unroll
  for (int i = 0; i < 2; ++i) {
    float4 u0 = ((const float4*)pA[i])[0];
    float4 u1 = ((const float4*)pA[i])[1];
    aReg[i][0] = u0.x; aReg[i][1] = u0.y; aReg[i][2] = u0.z; aReg[i][3] = u0.w;
    aReg[i][4] = u1.x; aReg[i][5] = u1.y; aReg[i][6] = u1.z; aReg[i][7] = u1.w;
#pragma unroll
    for (int j = 0; j < 8; ++j) bReg[i][j] = pB[i][(size_t)j * N];
  }

  v8f acc[2][4];
#pragma unroll
  for (int i = 0; i < 2; ++i)
#pragma unroll
    for (int j = 0; j < 4; ++j)
#pragma unroll
      for (int r = 0; r < 8; ++r) acc[i][j][r] = 0.0f;

  for (int k0 = 0; k0 < K; k0 += TK) {
    // commit staged registers to LDS (f32 -> bf16)
#pragma unroll
    for (int i = 0; i < 2; ++i) {
      v8bf av, bv;
#pragma unroll
      for (int j = 0; j < 8; ++j) {
        av[j] = (__bf16)aReg[i][j];
        bv[j] = (__bf16)bReg[i][j];
      }
      *(v8bf*)&lA [ar[i] * LPAD + acg[i]] = av;
      *(v8bf*)&lBt[bn[i] * LPAD + bkg[i]] = bv;
    }
    __syncthreads();

    // software pipeline: issue next tile's global loads over this tile's math
    if (k0 + TK < K) {
#pragma unroll
      for (int i = 0; i < 2; ++i) {
        pA[i] += TK;
        pB[i] += (size_t)TK * N;
        float4 u0 = ((const float4*)pA[i])[0];
        float4 u1 = ((const float4*)pA[i])[1];
        aReg[i][0] = u0.x; aReg[i][1] = u0.y; aReg[i][2] = u0.z; aReg[i][3] = u0.w;
        aReg[i][4] = u1.x; aReg[i][5] = u1.y; aReg[i][6] = u1.z; aReg[i][7] = u1.w;
#pragma unroll
        for (int j = 0; j < 8; ++j) bReg[i][j] = pB[i][(size_t)j * N];
      }
    }

    // fragment loads (ISA 7.12.2 layouts), all ds_load_b128
    v16bf aF[2];
#pragma unroll
    for (int tm = 0; tm < 2; ++tm) {
      int row = wm * 32 + tm * 16 + l16;
      v8bf a0 = *(const v8bf*)&lA[row * LPAD + half * 8];        // K 0..7 / 8..15
      v8bf a1 = *(const v8bf*)&lA[row * LPAD + 16 + half * 8];   // K 16..23 / 24..31
      aF[tm] = __builtin_shufflevector(a0, a1, 0,1,2,3,4,5,6,7,8,9,10,11,12,13,14,15);
    }
    v16bf bF[4];
#pragma unroll
    for (int tn = 0; tn < 4; ++tn) {
      int col = wn * 64 + tn * 16 + l16;
      v8bf b0 = *(const v8bf*)&lBt[col * LPAD + half * 8];       // K 0..7 / 8..15
      v8bf b1 = *(const v8bf*)&lBt[col * LPAD + 16 + half * 8];  // K 16..23 / 24..31
      bF[tn] = __builtin_shufflevector(b0, b1, 0,1,2,3,4,5,6,7,8,9,10,11,12,13,14,15);
    }
#pragma unroll
    for (int tm = 0; tm < 2; ++tm)
#pragma unroll
      for (int tn = 0; tn < 4; ++tn)
        acc[tm][tn] = __builtin_amdgcn_wmma_f32_16x16x32_bf16(
            false, aF[tm], false, bF[tn], (short)0, acc[tm][tn], false, false);
    __syncthreads();
  }

  // store D (C/D layout: VGPR r -> row r + 8*half, lane&15 -> col)
#pragma unroll
  for (int tm = 0; tm < 2; ++tm)
#pragma unroll
    for (int tn = 0; tn < 4; ++tn) {
      int col = colBase + wn * 64 + tn * 16 + l16;
#pragma unroll
      for (int r = 0; r < 8; ++r) {
        int row = rowBase + wm * 32 + tm * 16 + half * 8 + r;
        C[(size_t)row * N + col] = acc[tm][tn][r];
      }
    }
}

// ---------------- attention coefficient dot products -----------------------
// one wave per (node, head): as[n,h] = h.a_src, ad[n,h] = h.a_dst
__global__ void __launch_bounds__(256)
alpha_kernel(const float* __restrict__ h, const float* __restrict__ a_src,
             const float* __restrict__ a_dst, float* __restrict__ as,
             float* __restrict__ ad, int nNodes, int H, int C) {
  int g = blockIdx.x * 8 + (threadIdx.x >> 5);
  int lane = threadIdx.x & 31;
  if (g >= nNodes * H) return;
  int n = g / H, hh = g % H;
  const float* hp = h + (size_t)n * H * C + (size_t)hh * C;
  const float* sp = a_src + (size_t)hh * C;
  const float* dp = a_dst + (size_t)hh * C;
  float s0 = 0.f, s1 = 0.f;
  for (int c = lane; c < C; c += 32) {
    float v = hp[c];
    s0 += v * sp[c];
    s1 += v * dp[c];
  }
#pragma unroll
  for (int off = 16; off; off >>= 1) {
    s0 += __shfl_xor(s0, off, 32);
    s1 += __shfl_xor(s1, off, 32);
  }
  if (lane == 0) { as[g] = s0; ad[g] = s1; }
}

// ---------------- edge softmax ---------------------------------------------
__device__ __forceinline__ void edge_ends(const long long* ei, int e, int E,
                                          int& s, int& d) {
  if (e < E) { s = (int)ei[e]; d = (int)ei[E + e]; }
  else       { s = d = e - E; }            // self loops appended
}

__global__ void edge_logits(const long long* __restrict__ ei,
                            const float* __restrict__ as,
                            const float* __restrict__ ad,
                            float* __restrict__ e_out,
                            unsigned* __restrict__ mkey,
                            int E, int Etot, int H) {
  int i = blockIdx.x * blockDim.x + threadIdx.x;
  if (i >= Etot * H) return;
  int e = i / H, hh = i % H;
  int s, d; edge_ends(ei, e, E, s, d);
  float v = as[s * H + hh] + ad[d * H + hh];
  v = v > 0.f ? v : NEG_SLOPE * v;         // leaky relu
  e_out[i] = v;
  atomicMax(&mkey[d * H + hh], f2ord(v));  // segment max
}

__global__ void edge_softp(const long long* __restrict__ ei,
                           float* __restrict__ e_out,
                           const unsigned* __restrict__ mkey,
                           float* __restrict__ den,
                           int E, int Etot, int H) {
  int i = blockIdx.x * blockDim.x + threadIdx.x;
  if (i >= Etot * H) return;
  int e = i / H, hh = i % H;
  int s, d; edge_ends(ei, e, E, s, d);
  (void)s;
  float mv = ord2f(mkey[d * H + hh]);
  if (mv < -3.0e38f) mv = 0.f;             // isfinite guard
  float p = __expf(e_out[i] - mv);
  e_out[i] = p;
  atomicAdd(&den[d * H + hh], p);          // segment sum
}

// ---------------- weighted scatter-add of messages -------------------------
// one block per edge; out[dst, h*C+c] += h[src, h*C+c] * p/den
__global__ void __launch_bounds__(256)
aggregate(const long long* __restrict__ ei, const float* __restrict__ hsrc,
          const float* __restrict__ p, const float* __restrict__ den,
          float* __restrict__ out, int E, int H, int C) {
  int e = blockIdx.x;
  int s, d; edge_ends(ei, e, E, s, d);
  const float* hp = hsrc + (size_t)s * H * C;
  float* op = out + (size_t)d * H * C;
  for (int hh = 0; hh < H; ++hh) {
    float coef = p[e * H + hh] / den[d * H + hh];
    const float* hph = hp + hh * C;
    float* oph = op + hh * C;
    for (int c = threadIdx.x; c < C; c += blockDim.x)
      atomicAdd(&oph[c], hph[c] * coef);
  }
}

// ---------------- elementwise bias (+optional ELU) -------------------------
__global__ void bias_act(float* __restrict__ x, const float* __restrict__ b,
                         size_t n, int C, int do_elu) {
  size_t i = (size_t)blockIdx.x * blockDim.x + threadIdx.x;
  if (i >= n) return;
  float v = x[i] + b[i % (size_t)C];
  if (do_elu) v = v > 0.f ? v : (__expf(v) - 1.f);
  x[i] = v;
}

// ---------------- fills ----------------------------------------------------
__global__ void fill_f32(float* p, float v, size_t n) {
  size_t i = (size_t)blockIdx.x * blockDim.x + threadIdx.x;
  if (i < n) p[i] = v;
}
__global__ void fill_u32(unsigned* p, unsigned v, size_t n) {
  size_t i = (size_t)blockIdx.x * blockDim.x + threadIdx.x;
  if (i < n) p[i] = v;
}

// ---------------------------------------------------------------------------
extern "C" void kernel_launch(void* const* d_in, const int* in_sizes, int n_in,
                              void* d_out, int out_size, void* d_ws, size_t ws_size,
                              hipStream_t stream) {
  const float*     x      = (const float*)d_in[0];
  const long long* ei     = (const long long*)d_in[1];   // int64 [2, NE]
  const float*     W1     = (const float*)d_in[2];
  const float*     a_src1 = (const float*)d_in[3];
  const float*     a_dst1 = (const float*)d_in[4];
  const float*     b1     = (const float*)d_in[5];
  const float*     W2     = (const float*)d_in[6];
  const float*     a_src2 = (const float*)d_in[7];
  const float*     a_dst2 = (const float*)d_in[8];
  const float*     b2     = (const float*)d_in[9];
  float* out = (float*)d_out;

  // workspace layout (floats)
  float* fws = (float*)d_ws;
  size_t o = 0;
  float*    h1   = fws + o; o += (size_t)NN * HC1;   // pre-attn features L1
  float*    out1 = fws + o; o += (size_t)NN * HC1;   // aggregated L1
  float*    h2   = fws + o; o += (size_t)NN * DD;    // pre-attn features L2
  float*    as1  = fws + o; o += (size_t)NN * H1;
  float*    ad1  = fws + o; o += (size_t)NN * H1;
  float*    e1   = fws + o; o += (size_t)ET * H1;
  unsigned* m1   = (unsigned*)(fws + o); o += (size_t)NN * H1;
  float*    den1 = fws + o; o += (size_t)NN * H1;
  float*    as2  = fws + o; o += (size_t)NN;
  float*    ad2  = fws + o; o += (size_t)NN;
  float*    e2   = fws + o; o += (size_t)ET;
  unsigned* m2   = (unsigned*)(fws + o); o += (size_t)NN;
  float*    den2 = fws + o; o += (size_t)NN;

  auto cdiv = [](size_t a, size_t b) { return (unsigned)((a + b - 1) / b); };

  // ===== Layer 1 =====
  gemm_bf16_wmma<<<dim3(HC1 / TN, NN / TM), 256, 0, stream>>>(x, W1, h1, NN, HC1, DD);
  alpha_kernel<<<cdiv((size_t)NN * H1, 8), 256, 0, stream>>>(h1, a_src1, a_dst1,
                                                             as1, ad1, NN, H1, DD);
  fill_u32<<<cdiv((size_t)NN * H1, 256), 256, 0, stream>>>(m1, ORD_NEG_INF, (size_t)NN * H1);
  fill_f32<<<cdiv((size_t)NN * H1, 256), 256, 0, stream>>>(den1, 0.f, (size_t)NN * H1);
  fill_f32<<<cdiv((size_t)NN * HC1, 256), 256, 0, stream>>>(out1, 0.f, (size_t)NN * HC1);
  edge_logits<<<cdiv((size_t)ET * H1, 256), 256, 0, stream>>>(ei, as1, ad1, e1, m1, NE, ET, H1);
  edge_softp<<<cdiv((size_t)ET * H1, 256), 256, 0, stream>>>(ei, e1, m1, den1, NE, ET, H1);
  aggregate<<<ET, 256, 0, stream>>>(ei, h1, e1, den1, out1, NE, H1, DD);
  bias_act<<<cdiv((size_t)NN * HC1, 256), 256, 0, stream>>>(out1, b1, (size_t)NN * HC1, HC1, 1);

  // ===== Layer 2 =====
  gemm_bf16_wmma<<<dim3(DD / TN, NN / TM), 256, 0, stream>>>(out1, W2, h2, NN, DD, HC1);
  alpha_kernel<<<cdiv((size_t)NN, 8), 256, 0, stream>>>(h2, a_src2, a_dst2,
                                                        as2, ad2, NN, 1, DD);
  fill_u32<<<cdiv((size_t)NN, 256), 256, 0, stream>>>(m2, ORD_NEG_INF, (size_t)NN);
  fill_f32<<<cdiv((size_t)NN, 256), 256, 0, stream>>>(den2, 0.f, (size_t)NN);
  fill_f32<<<cdiv((size_t)NN * DD, 256), 256, 0, stream>>>(out, 0.f, (size_t)NN * DD);
  edge_logits<<<cdiv((size_t)ET, 256), 256, 0, stream>>>(ei, as2, ad2, e2, m2, NE, ET, 1);
  edge_softp<<<cdiv((size_t)ET, 256), 256, 0, stream>>>(ei, e2, m2, den2, NE, ET, 1);
  aggregate<<<ET, 256, 0, stream>>>(ei, h2, e2, den2, out, NE, 1, DD);
  bias_act<<<cdiv((size_t)NN * DD, 256), 256, 0, stream>>>(out, b2, (size_t)NN * DD, DD, 0);
}
